// HANLayer_13013750907160
// MI455X (gfx1250) — matrix operand order repaired and checked by reading, not compiled, not run on hardware
//
#include <hip/hip_runtime.h>

typedef __attribute__((ext_vector_type(16))) _Float16 v16h;
typedef __attribute__((ext_vector_type(8)))  _Float16 v8h;
typedef __attribute__((ext_vector_type(8)))  float    v8f;

namespace {
constexpr int N_   = 20000;
constexpr int E_   = 320000;
constexpr int M_   = 3;
constexpr int IN_  = 128;
constexpr int H_   = 4;
constexpr int D_   = 64;
constexpr int HD_  = 256;     // H_*D_
constexpr int HID_ = 128;
constexpr int FSTR_ = M_ * HD_;   // feat row stride: [N][M*HD]
constexpr float SLOPE = 0.2f;
}

#define USE_ASYNC_LDS 1

// ---------- helpers ----------
__device__ __forceinline__ float leaky(float v) { return v > 0.f ? v : SLOPE * v; }

__device__ __forceinline__ void atomicMaxF(float* addr, float val) {
    int cur = __float_as_int(*addr);
    while (val > __int_as_float(cur)) {
        int assumed = cur;
        cur = atomicCAS((int*)addr, assumed, __float_as_int(val));
        if (cur == assumed) break;
    }
}

// ---------- casts / transposes ----------
__global__ void k_cast_f16(const float* __restrict__ s, _Float16* __restrict__ d, int n) {
    for (int i = blockIdx.x * blockDim.x + threadIdx.x; i < n; i += gridDim.x * blockDim.x)
        d[i] = (_Float16)s[i];
}

// W: [M][IN][HD] (f32) -> Wt: [M*HD][IN] (f16): column (m*HD+c) is K-contiguous
__global__ void k_transpose_W(const float* __restrict__ W, _Float16* __restrict__ Wt) {
    const int n = M_ * IN_ * HD_;
    for (int i = blockIdx.x * blockDim.x + threadIdx.x; i < n; i += gridDim.x * blockDim.x) {
        int c = i % HD_;
        int t = i / HD_;
        int k = t % IN_;
        int m = t / IN_;
        Wt[((size_t)m * HD_ + c) * IN_ + k] = (_Float16)W[i];
    }
}

// sem_W1: [HD][HID] (f32) -> W1t: [HID][HD] (f16)
__global__ void k_transpose_sem(const float* __restrict__ W1, _Float16* __restrict__ W1t) {
    const int n = HD_ * HID_;
    for (int i = blockIdx.x * blockDim.x + threadIdx.x; i < n; i += gridDim.x * blockDim.x) {
        int c = i % HID_;
        int k = i / HID_;
        W1t[(size_t)c * HD_ + k] = (_Float16)W1[i];
    }
}

__global__ void k_fill(float* __restrict__ p, float v, int n) {
    for (int i = blockIdx.x * blockDim.x + threadIdx.x; i < n; i += gridDim.x * blockDim.x)
        p[i] = v;
}

// z[m][n][c] = bias[m][c]
__global__ void k_init_z(const float* __restrict__ bias, float* __restrict__ z) {
    const int n = M_ * N_ * HD_;
    for (int i = blockIdx.x * blockDim.x + threadIdx.x; i < n; i += gridDim.x * blockDim.x) {
        int c = i % HD_;
        int m = i / (N_ * HD_);
        z[i] = bias[m * HD_ + c];
    }
}

// ---------- generic WMMA GEMM: C[rows][cols] = A[rows][K] (f16) x Bt[cols][K] (f16) ----------
// Block = 256 threads = 8 waves; block covers one 16-row tile x 8 column tiles.
// A tile (16 x K f16) is staged once into LDS (async-to-LDS), shared by all 8 waves.
// LDS rows padded by 8 halves (272B stride) -> fragment ds_load_b128 is bank-conflict-free.
// Requires: K % 32 == 0, cols % 128 == 0, rows % 16 == 0, K <= 256.
__global__ void __launch_bounds__(256)
k_gemm_wmma(const _Float16* __restrict__ A,
            const _Float16* __restrict__ Bt,
            float* __restrict__ C,
            int K, int cols)
{
    constexpr int PAD = 8;                       // halves
    __shared__ __align__(16) _Float16 As[16 * (256 + PAD)];

    const int lane = threadIdx.x & 31;
    const int wave = threadIdx.x >> 5;
    const int cgroups = cols >> 7;               // groups of 8 column tiles
    const int tr  = blockIdx.x / cgroups;
    const int tc  = (blockIdx.x % cgroups) * 8 + wave;
    const int l15  = lane & 15;
    const int half = lane >> 4;
    const int ldsK = K + PAD;

    // ---- stage A tile: 16 rows x K halves, 16B segments per thread ----
    const _Float16* atile = A + (size_t)tr * 16 * K;
    const int segsPerRow = K >> 3;               // 16B = 8 halves per segment
#if USE_ASYNC_LDS
    {
        unsigned lbase = (unsigned)(size_t)(void*)As;
        for (int sg = threadIdx.x; sg < 16 * segsPerRow; sg += 256) {
            int r = sg / segsPerRow, s = sg - r * segsPerRow;
            unsigned laddr = lbase + (unsigned)((r * ldsK + s * 8) * 2);
            const _Float16* g = atile + (size_t)r * K + s * 8;
            asm volatile("global_load_async_to_lds_b128 %0, %1, off"
                         :: "v"(laddr), "v"(g) : "memory");
        }
        asm volatile("s_wait_asynccnt 0x0" ::: "memory");
    }
#else
    for (int sg = threadIdx.x; sg < 16 * segsPerRow; sg += 256) {
        int r = sg / segsPerRow, s = sg - r * segsPerRow;
        *(v8h*)(As + r * ldsK + s * 8) = *(const v8h*)(atile + (size_t)r * K + s * 8);
    }
#endif
    __syncthreads();

    const _Float16* arow = As + l15 * ldsK;                       // LDS, padded row
    const _Float16* bcol = Bt + (size_t)(tc * 16 + l15) * K;      // global, K-contiguous

    v8f acc = {};
    for (int k0 = 0; k0 < K; k0 += 32) {
        // A fragment: e=0..7 -> K=k0+8*half+e ; e=8..15 -> K=k0+16+8*half+(e-8)
        v8h alo = *(const v8h*)(arow + k0 + half * 8);
        v8h ahi = *(const v8h*)(arow + k0 + 16 + half * 8);
        v16h a, b;
#pragma unroll
        for (int i = 0; i < 8; ++i) { a[i] = alo[i]; a[i + 8] = ahi[i]; }
        // B fragment: e=0..7 -> K=k0+8*half+e ; e=8..15 -> K=k0+16+8*half+(e-8) at fixed column
        v8h blo = *(const v8h*)(bcol + k0 + half * 8);
        v8h bhi = *(const v8h*)(bcol + k0 + 16 + half * 8);
#pragma unroll
        for (int i = 0; i < 8; ++i) { b[i] = blo[i]; b[i + 8] = bhi[i]; }
        acc = __builtin_amdgcn_wmma_f32_16x16x32_f16(
            false, a, false, b, (short)0, acc, false, false);
    }
    // C/D layout: VGPR j -> row j+8*half, col l15
    float* crow = C + (size_t)(tr * 16) * cols + tc * 16;
#pragma unroll
    for (int j = 0; j < 8; ++j)
        crow[(size_t)(j + 8 * half) * cols + l15] = acc[j];
}

// ---------- per-node attention logits (feat layout [N][M*HD]) ----------
__global__ void k_elr(const float* __restrict__ feat,
                      const float* __restrict__ al,   // [M][H][D]
                      const float* __restrict__ ar,
                      float* __restrict__ el, float* __restrict__ er) {
    const int n = M_ * N_ * H_;
    for (int i = blockIdx.x * blockDim.x + threadIdx.x; i < n; i += gridDim.x * blockDim.x) {
        int h = i & (H_ - 1);
        int t = i >> 2;               // m*N_ + node
        int node = t % N_;
        int m = t / N_;
        const float4* f = (const float4*)(feat + (size_t)node * FSTR_ + m * HD_ + h * D_);
        const float4* a = (const float4*)(al + (size_t)(m * H_ + h) * D_);
        const float4* b = (const float4*)(ar + (size_t)(m * H_ + h) * D_);
        float sl = 0.f, sr = 0.f;
#pragma unroll
        for (int j = 0; j < D_ / 4; ++j) {
            float4 fv = f[j], av = a[j], bv = b[j];
            sl += fv.x * av.x + fv.y * av.y + fv.z * av.z + fv.w * av.w;
            sr += fv.x * bv.x + fv.y * bv.y + fv.z * bv.z + fv.w * bv.w;
        }
        el[i] = sl; er[i] = sr;
    }
}

// ---------- edge softmax ----------
__global__ void k_edge_logits(const int* __restrict__ src, const int* __restrict__ dst,
                              const float* __restrict__ el, const float* __restrict__ er,
                              float* __restrict__ elog, float* __restrict__ emax) {
    const int n = M_ * E_;
    for (int i = blockIdx.x * blockDim.x + threadIdx.x; i < n; i += gridDim.x * blockDim.x) {
        int m = i / E_;
        int s = src[i], d = dst[i];
        const float* pl = el + (size_t)(m * N_ + s) * H_;
        const float* pr = er + (size_t)(m * N_ + d) * H_;
        float* pe = elog + (size_t)i * H_;
        float* pm = emax + (size_t)(m * N_ + d) * H_;
#pragma unroll
        for (int h = 0; h < H_; ++h) {
            float v = leaky(pl[h] + pr[h]);
            pe[h] = v;
            atomicMaxF(pm + h, v);
        }
    }
}

__global__ void k_edge_exp(const int* __restrict__ dst,
                           const float* __restrict__ emax,
                           float* __restrict__ elog, float* __restrict__ denom) {
    const int n = M_ * E_;
    for (int i = blockIdx.x * blockDim.x + threadIdx.x; i < n; i += gridDim.x * blockDim.x) {
        int m = i / E_;
        int d = dst[i];
        const float* pm = emax + (size_t)(m * N_ + d) * H_;
        float* pe = elog + (size_t)i * H_;
        float* pd = denom + (size_t)(m * N_ + d) * H_;
#pragma unroll
        for (int h = 0; h < H_; ++h) {
            float ee = expf(pe[h] - pm[h]);
            pe[h] = ee;
            atomicAdd(pd + h, ee);
        }
    }
}

// one block (256 threads) per edge: thread c handles output channel c, head c>>6
__global__ void __launch_bounds__(256)
k_edge_scatter(const int* __restrict__ src, const int* __restrict__ dst,
               const float* __restrict__ elog, const float* __restrict__ denom,
               const float* __restrict__ feat, float* __restrict__ z) {
    const int i = blockIdx.x;          // [0, M_*E_)
    const int m = i / E_;
    const int s = src[i], d = dst[i];
    const int c = threadIdx.x;
    const int h = c >> 6;
    float ee = elog[(size_t)i * H_ + h];
    float dn = denom[(size_t)(m * N_ + d) * H_ + h];
    float alpha = ee / fmaxf(dn, 1e-9f);
    float v = alpha * feat[(size_t)s * FSTR_ + m * HD_ + c];
    atomicAdd(&z[(size_t)(m * N_ + d) * HD_ + c], v);
}

// ---------- semantic attention tail: w = tanh(p + b1) @ W2, mean-accumulate ----------
__global__ void __launch_bounds__(128)
k_semfin(const float* __restrict__ p, const float* __restrict__ b1,
         const float* __restrict__ W2, float* __restrict__ wsum) {
    __shared__ float red[HID_];
    const int row = blockIdx.x;        // m*N_ + n
    const int m = row / N_;
    const int t = threadIdx.x;
    red[t] = tanhf(p[(size_t)row * HID_ + t] + b1[t]) * W2[t];
    __syncthreads();
    for (int off = 64; off > 0; off >>= 1) {
        if (t < off) red[t] += red[t + off];
        __syncthreads();
    }
    if (t == 0) atomicAdd(&wsum[m], red[0]);
}

__global__ void k_beta(const float* __restrict__ wsum, float* __restrict__ beta) {
    if (threadIdx.x == 0 && blockIdx.x == 0) {
        float v0 = wsum[0] / (float)N_, v1 = wsum[1] / (float)N_, v2 = wsum[2] / (float)N_;
        float mx = fmaxf(v0, fmaxf(v1, v2));
        float e0 = expf(v0 - mx), e1 = expf(v1 - mx), e2 = expf(v2 - mx);
        float s = e0 + e1 + e2;
        beta[0] = e0 / s; beta[1] = e1 / s; beta[2] = e2 / s;
    }
}

__global__ void k_combine(const float* __restrict__ z, const float* __restrict__ beta,
                          float* __restrict__ out) {
    const int n = N_ * HD_;
    const float b0 = beta[0], b1 = beta[1], b2 = beta[2];
    for (int i = blockIdx.x * blockDim.x + threadIdx.x; i < n; i += gridDim.x * blockDim.x)
        out[i] = b0 * z[i] + b1 * z[(size_t)N_ * HD_ + i] + b2 * z[2 * (size_t)N_ * HD_ + i];
}

// ---------- launch ----------
extern "C" void kernel_launch(void* const* d_in, const int* in_sizes, int n_in,
                              void* d_out, int out_size, void* d_ws, size_t ws_size,
                              hipStream_t stream) {
    (void)in_sizes; (void)n_in; (void)out_size; (void)ws_size;
    const float* h    = (const float*)d_in[0];
    const int*   src  = (const int*)d_in[1];
    const int*   dst  = (const int*)d_in[2];
    const float* W    = (const float*)d_in[3];
    const float* al   = (const float*)d_in[4];
    const float* ar   = (const float*)d_in[5];
    const float* bias = (const float*)d_in[6];
    const float* sW1  = (const float*)d_in[7];
    const float* sb1  = (const float*)d_in[8];
    const float* sW2  = (const float*)d_in[9];
    float* out = (float*)d_out;

    char* wsp = (char*)d_ws;
    auto alloc = [&](size_t bytes) {
        char* p = wsp;
        wsp += (bytes + 255) & ~(size_t)255;
        return p;
    };
    _Float16* h16   = (_Float16*)alloc((size_t)N_ * IN_ * 2);
    _Float16* Wt16  = (_Float16*)alloc((size_t)M_ * HD_ * IN_ * 2);
    _Float16* W1t16 = (_Float16*)alloc((size_t)HID_ * HD_ * 2);
    float* feat  = (float*)alloc((size_t)N_ * FSTR_ * 4);   // [N][M*HD]; reused below
    float* z     = (float*)alloc((size_t)M_ * N_ * HD_ * 4);
    float* el    = (float*)alloc((size_t)M_ * N_ * H_ * 4);
    float* er    = (float*)alloc((size_t)M_ * N_ * H_ * 4);
    float* emax  = (float*)alloc((size_t)M_ * N_ * H_ * 4);
    float* denom = (float*)alloc((size_t)M_ * N_ * H_ * 4);
    float* elog  = (float*)alloc((size_t)M_ * E_ * H_ * 4);
    float* wsum  = (float*)alloc(64);
    float* beta  = (float*)alloc(64);
    // feat (61.44 MB) is dead after k_edge_scatter; alias z16 (30.72 MB) + p (30.72 MB) into it.
    _Float16* z16 = (_Float16*)feat;
    float*    p   = (float*)((char*)feat + (size_t)M_ * N_ * HD_ * 2);

    // ---- feature GEMM: feat[n][m*HD+c] = h @ W  (rows=N, K=128, cols=768) ----
    k_cast_f16<<<2048, 256, 0, stream>>>(h, h16, N_ * IN_);
    k_transpose_W<<<384, 256, 0, stream>>>(W, Wt16);
    k_gemm_wmma<<<(N_ / 16) * (FSTR_ / 128), 256, 0, stream>>>(h16, Wt16, feat, IN_, FSTR_);

    // ---- GAT per metapath ----
    k_elr<<<960, 256, 0, stream>>>(feat, al, ar, el, er);
    k_fill<<<960, 256, 0, stream>>>(emax, -__builtin_inff(), M_ * N_ * H_);
    k_fill<<<960, 256, 0, stream>>>(denom, 0.f, M_ * N_ * H_);
    k_fill<<<1, 64, 0, stream>>>(wsum, 0.f, M_);
    k_init_z<<<4096, 256, 0, stream>>>(bias, z);
    k_edge_logits<<<3750, 256, 0, stream>>>(src, dst, el, er, elog, emax);
    k_edge_exp<<<3750, 256, 0, stream>>>(dst, emax, elog, denom);
    k_edge_scatter<<<M_ * E_, 256, 0, stream>>>(src, dst, elog, denom, feat, z);

    // ---- semantic attention: p = z16 @ W1t (rows=M*N, K=256, cols=128) ----
    k_cast_f16<<<4096, 256, 0, stream>>>(z, z16, M_ * N_ * HD_);
    k_transpose_sem<<<128, 256, 0, stream>>>(sW1, W1t16);
    k_gemm_wmma<<<(M_ * N_ / 16) * (HID_ / 128), 256, 0, stream>>>(z16, W1t16, p, HD_, HID_);
    k_semfin<<<M_ * N_, 128, 0, stream>>>(p, sb1, sW2, wsum);
    k_beta<<<1, 32, 0, stream>>>(wsum, beta);
    k_combine<<<5000, 256, 0, stream>>>(z, beta, out);
}
